// Attention_10943576670580
// MI455X (gfx1250) — matrix operand order repaired and checked by reading, not compiled, not run on hardware
//
#include <hip/hip_runtime.h>

#define HID 512
#define K2H 1024
#define SEQ 2048
#define NB  32
#define MT  64

#define ABYTES (64 * 1024 * 2)                 // A tile: 64x1024 bf16 = 128 KB
#define BBYTES (32 * 1024 * 2)                 // B tile: 32x1024 bf16 = 64 KB
#define SMEM_BYTES (ABYTES + 2 * BBYTES + 64 * 4)   // A + B ping-pong + scores

typedef __bf16 bf16_t;
typedef __attribute__((ext_vector_type(16))) bf16_t       v16bf;
typedef __attribute__((ext_vector_type(8)))  float        v8f;
typedef __attribute__((ext_vector_type(4)))  unsigned int v4u;
typedef __attribute__((ext_vector_type(2)))  unsigned int v2u;

union FragU { v4u q[2]; v16bf h; };

__device__ __forceinline__ unsigned int f2bf(float f) {
  unsigned int u = __float_as_uint(f);
  u += 0x7FFFu + ((u >> 16) & 1u);   // round-to-nearest-even
  return u >> 16;
}
__device__ __forceinline__ unsigned int pk2(float a, float b) {
  return f2bf(a) | (f2bf(b) << 16);
}

// Branch-free tanh: one v_exp_f32 (TRANS, co-executes with WMMA) + rcp.
__device__ __forceinline__ float fast_tanh(float x) {
  float cx = fminf(fmaxf(x, -9.0f), 9.0f);
  float e  = __expf(2.0f * cx);
  return 1.0f - __fdividef(2.0f, e + 1.0f);
}

// Async DMA one 64KB We tile (32 rows x 1024 bf16) into LDS buffer `buf`.
// GVS mode: global_load_async_to_lds_b128 vdst(LDS byte addr), vaddr(u32 off),
// saddr(64-bit base). Tracked by ASYNCcnt.
__device__ __forceinline__ void stage_b_async(const unsigned short* WeBf,
                                              int np, int buf, int tid) {
  unsigned long long gbase =
      (unsigned long long)(WeBf + (size_t)np * 32 * 1024);
  unsigned lds0 = (unsigned)(ABYTES + buf * BBYTES + tid * 16);
  unsigned goff = (unsigned)(tid * 16);
#pragma unroll
  for (int it = 0; it < 16; ++it) {   // 256 thr x 16B x 16 = 64 KB
    asm volatile("global_load_async_to_lds_b128 %0, %1, %2"
                 :
                 : "v"(lds0 + (unsigned)(it * 4096)),
                   "v"(goff + (unsigned)(it * 4096)), "s"(gbase)
                 : "memory");
  }
}

__device__ __forceinline__ void wait_async0() {
  asm volatile("s_wait_asynccnt 0x0" ::: "memory");
}

// ---------------------------------------------------------------------------
// Kernel 1: convert We = attn_W[:, 512:1536] (f32, row stride 1536) into a
// contiguous bf16 matrix [512][1024] in workspace.
// ---------------------------------------------------------------------------
__global__ void __launch_bounds__(256) prep_we_kernel(
    const float* __restrict__ attn_W, unsigned short* __restrict__ WeBf) {
  int i = (blockIdx.x * 256 + threadIdx.x) * 4;
  int o = i >> 10;
  int e = i & 1023;
  float4 w = *(const float4*)(attn_W + (size_t)o * 1536 + 512 + e);
  v2u p; p.x = pk2(w.x, w.y); p.y = pk2(w.z, w.w);
  *(v2u*)(WeBf + i) = p;
}

// ---------------------------------------------------------------------------
// Kernel 2: base[b][o] = hidden[b,:] . attn_W[o, 0:512] + attn_b[o]
// ---------------------------------------------------------------------------
__global__ void __launch_bounds__(256) hproj_kernel(
    const float* __restrict__ hidden, const float* __restrict__ attn_W,
    const float* __restrict__ attn_b, float* __restrict__ basev) {
  int idx = blockIdx.x * 256 + threadIdx.x;
  int b = idx >> 9, o = idx & 511;
  const float4* h4 = (const float4*)(hidden + (size_t)b * HID);
  const float4* w4 = (const float4*)(attn_W + (size_t)o * 1536);
  float acc = attn_b[o];
#pragma unroll 4
  for (int k = 0; k < HID / 4; ++k) {
    float4 h = h4[k], w = w4[k];
    acc += h.x * w.x + h.y * w.y + h.z * w.z + h.w * w.w;
  }
  basev[idx] = acc;
}

// ---------------------------------------------------------------------------
// Kernel 3: fused e_proj GEMM (bf16 WMMA, f32 accum) + tanh + v-dot.
// 8 waves; A tile staged once (f32->bf16), B tiles double-buffered via
// global_load_async_to_lds_b128 overlapped with WMMA compute.
// ---------------------------------------------------------------------------
__global__ void __launch_bounds__(256) energy_kernel(
    const float* __restrict__ enc, const unsigned short* __restrict__ WeBf,
    const float* __restrict__ basev, const float* __restrict__ vW,
    float* __restrict__ scores) {
  extern __shared__ char smem[];
  unsigned short* Ash = (unsigned short*)smem;                 // [64][1024]
  unsigned short* Bsh = (unsigned short*)(smem + ABYTES);      // 2x[32][1024]
  float* ssc = (float*)(smem + ABYTES + 2 * BBYTES);           // [64]

  const int tid = threadIdx.x;
  const int b   = blockIdx.y;
  const int m0  = blockIdx.x * MT;

  if (tid < MT) ssc[tid] = 0.0f;

  // Kick off async DMA of the first We tile; overlaps with A conversion below.
  stage_b_async(WeBf, 0, 0, tid);

  // Stage encoder tile: 64 contiguous rows, f32 -> bf16 into LDS
  {
    const float4* src = (const float4*)(enc + ((size_t)b * SEQ + m0) * K2H);
    v2u* dst = (v2u*)Ash;
#pragma unroll 4
    for (int i = tid; i < 64 * 1024 / 4; i += 256) {
      float4 v = src[i];
      v2u p; p.x = pk2(v.x, v.y); p.y = pk2(v.z, v.w);
      dst[i] = p;
    }
  }
  wait_async0();
  __syncthreads();

  const int wave = tid >> 5;
  const int lane = tid & 31;
  const int g    = lane >> 4;   // lane group (K split per ISA layouts)
  const int ln   = lane & 15;
  const int msub = wave & 3;
  const int nloc = wave >> 2;

  const unsigned short* Arow = Ash + (msub * 16 + ln) * 1024;

  float part[8];
#pragma unroll
  for (int r = 0; r < 8; ++r) part[r] = 0.0f;

  for (int np = 0; np < 16; ++np) {
    const int cur = np & 1;
    if (np < 15) stage_b_async(WeBf, np + 1, cur ^ 1, tid);  // prefetch DMA

    const unsigned short* Brow =
        Bsh + cur * 32 * 1024 + (nloc * 16 + ln) * 1024;

    const int   o  = np * 32 + nloc * 16 + ln;   // this lane's output column
    const float cb = basev[(b << 9) + o];        // h_proj + bias
    const float vw = vW[o];
    v8f c = {cb, cb, cb, cb, cb, cb, cb, cb};

#pragma unroll 8
    for (int kk = 0; kk < 32; ++kk) {
      FragU a, bb;
      // A (16-bit 16x32 layout): lanes 0-15 hold K {0..7,16..23}, 16-31 hold
      // K {8..15,24..31} -> two 16B LDS loads per lane
      a.q[0] = *(const v4u*)(Arow + kk * 32 + g * 8);
      a.q[1] = *(const v4u*)(Arow + kk * 32 + 16 + g * 8);
      // B (32x16): lane group g holds K {g*8..g*8+7, 16+g*8..16+g*8+7}
      bb.q[0] = *(const v4u*)(Brow + kk * 32 + g * 8);
      bb.q[1] = *(const v4u*)(Brow + kk * 32 + 16 + g * 8);
      c = __builtin_amdgcn_wmma_f32_16x16x32_bf16(false, a.h, false, bb.h,
                                                  (short)0, c, false, false);
    }

    // energy -> tanh -> weight by v[o]; accumulate per-row partial scores.
    // C/D layout: VGPR r holds row M = r + 8*g, column N = ln.
#pragma unroll
    for (int r = 0; r < 8; ++r) part[r] += fast_tanh(c[r]) * vw;

    wait_async0();      // next buffer's DMA complete before anyone reads it
    __syncthreads();
  }

#pragma unroll
  for (int r = 0; r < 8; ++r)
    atomicAdd(&ssc[msub * 16 + g * 8 + r], part[r]);
  __syncthreads();

  if (tid < MT) scores[(size_t)b * SEQ + m0 + tid] = ssc[tid];
}

// ---------------------------------------------------------------------------
// Kernel 4: row softmax over S=2048, one block per batch.
// ---------------------------------------------------------------------------
__global__ void __launch_bounds__(256) softmax_kernel(
    const float* __restrict__ scores, float* __restrict__ out) {
  __shared__ float red[256];
  const int b = blockIdx.x, t = threadIdx.x;
  const float* row = scores + (size_t)b * SEQ;
  float v[8], mx = -3.4e38f;
#pragma unroll
  for (int i = 0; i < 8; ++i) { v[i] = row[t + i * 256]; mx = fmaxf(mx, v[i]); }
  red[t] = mx; __syncthreads();
  for (int s = 128; s > 0; s >>= 1) {
    if (t < s) red[t] = fmaxf(red[t], red[t + s]);
    __syncthreads();
  }
  mx = red[0]; __syncthreads();
  float sum = 0.f;
#pragma unroll
  for (int i = 0; i < 8; ++i) { v[i] = expf(v[i] - mx); sum += v[i]; }
  red[t] = sum; __syncthreads();
  for (int s = 128; s > 0; s >>= 1) {
    if (t < s) red[t] += red[t + s];
    __syncthreads();
  }
  const float inv = 1.0f / red[0];
#pragma unroll
  for (int i = 0; i < 8; ++i) out[(size_t)b * SEQ + t + i * 256] = v[i] * inv;
}

// ---------------------------------------------------------------------------
extern "C" void kernel_launch(void* const* d_in, const int* in_sizes, int n_in,
                              void* d_out, int out_size, void* d_ws,
                              size_t ws_size, hipStream_t stream) {
  const float* hidden = (const float*)d_in[0];   // (32, 512)
  const float* enc    = (const float*)d_in[1];   // (32, 2048, 1024)
  const float* attn_W = (const float*)d_in[2];   // (512, 1536)
  const float* attn_b = (const float*)d_in[3];   // (512,)
  const float* vW     = (const float*)d_in[4];   // (1, 512)
  float* out = (float*)d_out;                    // (32, 2048)

  unsigned short* WeBf = (unsigned short*)d_ws;                        // 1 MiB
  float* basev  = (float*)((char*)d_ws + (1u << 20));                  // 64 KiB
  float* scores = (float*)((char*)d_ws + (1u << 20) + (64u << 10));    // 256 KiB

  prep_we_kernel<<<512, 256, 0, stream>>>(attn_W, WeBf);
  hproj_kernel<<<64, 256, 0, stream>>>(hidden, attn_W, attn_b, basev);
  energy_kernel<<<dim3(SEQ / MT, NB), 256, SMEM_BYTES, stream>>>(
      enc, WeBf, basev, vW, scores);
  softmax_kernel<<<NB, 256, 0, stream>>>(scores, out);
}
